// SelfAttention3D_60928406061685
// MI455X (gfx1250) — compile-verified, matrix operand order checked by reading, not run on hardware
//
#include <hip/hip_runtime.h>

#define IN_DIM 64
#define QK_DIM 8
#define NB 2
#define NVOX 8000                 // 20*20*20

#define MWAVE 16                  // queries per wave (one WMMA column tile)
#define WAVES 4
#define MBLOCK (MWAVE * WAVES)    // 64 queries per block; 8000 = 125 * 64 exactly
#define NT 32                     // keys per streamed chunk; 8000 = 250 * 32 exactly
#define TILES_PER_B (NVOX / MBLOCK)

static_assert(NVOX % MBLOCK == 0, "M tiling must divide N");
static_assert(NVOX % NT == 0, "key tiling must divide N");

typedef __attribute__((ext_vector_type(2)))  float    v2f;
typedef __attribute__((ext_vector_type(8)))  float    v8f;
typedef __attribute__((ext_vector_type(8)))  _Float16 v8h;
typedef __attribute__((ext_vector_type(16))) _Float16 v16h;

// ---------------------------------------------------------------------------
// Kernel 1: fused 1x1x1-conv projections.
//   qt, kt : fp32 [B][N][8]   (row-contiguous -> WMMA f32 A/B operands are b64 loads)
//   vh     : f16  [B][64][N]  (key-dim contiguous -> WMMA f16 A operands are b128 loads)
// ---------------------------------------------------------------------------
__global__ __launch_bounds__(256) void qkv_proj_kernel(
    const float* __restrict__ x,
    const float* __restrict__ wq, const float* __restrict__ bq,
    const float* __restrict__ wk, const float* __restrict__ bk,
    const float* __restrict__ wv, const float* __restrict__ bv,
    float* __restrict__ qt, float* __restrict__ kt, _Float16* __restrict__ vh)
{
  const int b = blockIdx.y;
  const int n = blockIdx.x * blockDim.x + threadIdx.x;
  if (n >= NVOX) return;

  const float* xb = x + (size_t)b * IN_DIM * NVOX;
  float xv[IN_DIM];
#pragma unroll
  for (int c = 0; c < IN_DIM; ++c) xv[c] = xb[(size_t)c * NVOX + n];   // coalesced

  float* qn = qt + ((size_t)b * NVOX + n) * QK_DIM;
  float* kn = kt + ((size_t)b * NVOX + n) * QK_DIM;
#pragma unroll
  for (int o = 0; o < QK_DIM; ++o) {
    float aq = bq[o], ak = bk[o];
#pragma unroll
    for (int c = 0; c < IN_DIM; ++c) {
      aq = fmaf(wq[o * IN_DIM + c], xv[c], aq);   // uniform index -> scalar loads
      ak = fmaf(wk[o * IN_DIM + c], xv[c], ak);
    }
    qn[o] = aq;
    kn[o] = ak;
  }
  for (int o = 0; o < IN_DIM; ++o) {
    float av = bv[o];
#pragma unroll
    for (int c = 0; c < IN_DIM; ++c) av = fmaf(wv[o * IN_DIM + c], xv[c], av);
    vh[((size_t)b * IN_DIM + o) * NVOX + n] = (_Float16)av;            // coalesced
  }
}

// ---------------------------------------------------------------------------
// Kernel 2: flash-attention stream, fully transposed formulation.
// One wave = 16 queries (always the WMMA *column* dimension -> query stats are
// per-lane everywhere; no cross-lane broadcasts, no LDS).
//   S^T = K Q^T : fp32 WMMA 16x16x4 (K-dim 8 = 2 chained), 2 key tiles.
//     C-layout: lane L holds keys r+8g (+16 tile 1) of query column ln.
//     -> softmax reduction is in-lane + one xor-16 combine.
//   P^T = exp(S^T - m) packed f16 in registers == the B-operand layout.
//   O^T += V P^T : f16 WMMA 16x16x32 (V tile = documented A-operand layout),
//     4 row tiles cover C=64.  O^T C-layout: row c = r+8g(+16t), column = query ln
//     -> rescale and final 1/rowsum are per-lane; epilogue stores coalesced.
// ---------------------------------------------------------------------------
__global__ __launch_bounds__(MBLOCK * 2) void attn_stream_kernel(
    const float* __restrict__ qt, const float* __restrict__ kt,
    const _Float16* __restrict__ vh,
    const float* __restrict__ x, const float* __restrict__ gamma,
    float* __restrict__ out)
{
  const int lane = threadIdx.x & 31;
  const int wave = threadIdx.x >> 5;
  const int g    = lane >> 4;       // half-wave group (selects K pair / M block)
  const int ln   = lane & 15;
  const int b    = blockIdx.x / TILES_PER_B;
  const int tile = blockIdx.x % TILES_PER_B;
  const int row0 = tile * MBLOCK + wave * MWAVE;

  const float*    qb = qt + (size_t)b * NVOX * QK_DIM;
  const float*    kb = kt + (size_t)b * NVOX * QK_DIM;
  const _Float16* vb = vh + (size_t)b * IN_DIM * NVOX;

  // Q as fp32 16x16x4 B operand: lane ln -> query column, VGPR pair K = {2g,2g+1}.
  const int qm  = row0 + ln;
  const v2f qb0 = *(const v2f*)(qb + (size_t)qm * QK_DIM + 2 * g);       // K 0..3
  const v2f qb1 = *(const v2f*)(qb + (size_t)qm * QK_DIM + 4 + 2 * g);   // K 4..7

  // Running softmax state for THIS lane's query (lanes ln and ln+16 mirror).
  float lmax = -3.0e38f;
  float lsum = 0.0f;
  v8f oacc[4] = {v8f{}, v8f{}, v8f{}, v8f{}};   // O^T: c-tile t, row r+8g, col ln

  for (int n0 = 0; n0 < NVOX; n0 += NT) {
    // ---- S^T = K Q^T : 4x V_WMMA_F32_16X16X4_F32 (fp32 logits) ----
    v8f st0 = {}, st1 = {};
    {
      const size_t a0 = (size_t)(n0 + ln) * QK_DIM;          // key tile 0, row ln
      const v2f ka0 = *(const v2f*)(kb + a0 + 2 * g);
      const v2f ka1 = *(const v2f*)(kb + a0 + 4 + 2 * g);
      st0 = __builtin_amdgcn_wmma_f32_16x16x4_f32(false, ka0, false, qb0, (short)0, st0, false, false);
      st0 = __builtin_amdgcn_wmma_f32_16x16x4_f32(false, ka1, false, qb1, (short)0, st0, false, false);
      const size_t a1 = (size_t)(n0 + 16 + ln) * QK_DIM;     // key tile 1
      const v2f ka2 = *(const v2f*)(kb + a1 + 2 * g);
      const v2f ka3 = *(const v2f*)(kb + a1 + 4 + 2 * g);
      st1 = __builtin_amdgcn_wmma_f32_16x16x4_f32(false, ka2, false, qb0, (short)0, st1, false, false);
      st1 = __builtin_amdgcn_wmma_f32_16x16x4_f32(false, ka3, false, qb1, (short)0, st1, false, false);
    }

    // ---- chunk max for this lane's query: in-lane tree + one xor-16 combine ----
    float cm = fmaxf(st0[0], st1[0]);
#pragma unroll
    for (int r = 1; r < 8; ++r) cm = fmaxf(cm, fmaxf(st0[r], st1[r]));
    cm = fmaxf(cm, __shfl_xor(cm, 16, 32));
    const float mn = fmaxf(lmax, cm);

    // ---- P^T = exp(S^T - max), packed f16 directly into the PV B-operand ----
    // B-operand (16-bit 32x16): lane L column ln, elems 0..7 -> K = 8g..8g+7,
    // elems 8..15 -> K = 16+8g..16+8g+7.  st0[r] is key r+8g, st1[r] key 16+r+8g.
    union { v16h v; _Float16 e[16]; } pt;
    float cs = 0.0f;
#pragma unroll
    for (int r = 0; r < 8; ++r) {
      const float p0 = __expf(st0[r] - mn);
      const float p1 = __expf(st1[r] - mn);
      pt.e[r]     = (_Float16)p0;
      pt.e[8 + r] = (_Float16)p1;
      cs += p0 + p1;
    }
    cs += __shfl_xor(cs, 16, 32);
    const float sc = __expf(lmax - mn);                  // 0 on first chunk
    lsum = lsum * sc + cs;
    lmax = mn;

    // ---- rescale O^T: columns are queries -> pure per-lane multiply ----
#pragma unroll
    for (int t = 0; t < 4; ++t)
#pragma unroll
      for (int r = 0; r < 8; ++r) oacc[t][r] *= sc;

    // ---- O^T += V P^T : 4x V_WMMA_F32_16X16X32_F16 over C row tiles ----
    // V tile as A operand (documented layout): lane row c = 16t+ln,
    // elems 0..7 -> keys n0+8g.., elems 8..15 -> keys n0+16+8g..
#pragma unroll
    for (int t = 0; t < 4; ++t) {
      union { v16h v; v8h h[2]; } vA;
      const _Float16* vr = vb + (size_t)(t * 16 + ln) * NVOX + n0;
      vA.h[0] = *(const v8h*)(vr + 8 * g);
      vA.h[1] = *(const v8h*)(vr + 16 + 8 * g);
      oacc[t] = __builtin_amdgcn_wmma_f32_16x16x32_f16(
          false, vA.v, false, pt.v, (short)0, oacc[t], false, false);
    }

    if (n0 + NT < NVOX) {
      __builtin_prefetch(kb + (size_t)(n0 + NT + ln) * QK_DIM, 0, 0);
      __builtin_prefetch(vb + (size_t)ln * NVOX + n0 + NT, 0, 0);
    }
  }

  // ---- epilogue: out = gamma * (O^T / rowsum) + x  (coalesced stores) ----
  const float gm  = gamma[0];
  const float is  = 1.0f / lsum;                         // per-query, this lane
  const float* xb = x + (size_t)b * IN_DIM * NVOX;
  float* ob = out + (size_t)b * IN_DIM * NVOX;
#pragma unroll
  for (int t = 0; t < 4; ++t)
#pragma unroll
    for (int r = 0; r < 8; ++r) {
      const int    c   = t * 16 + r + 8 * g;             // channel (O^T row)
      const size_t idx = (size_t)c * NVOX + row0 + ln;   // lanes -> consecutive voxels
      ob[idx] = gm * (oacc[t][r] * is) + xb[idx];
    }
}

// ---------------------------------------------------------------------------
extern "C" void kernel_launch(void* const* d_in, const int* in_sizes, int n_in,
                              void* d_out, int out_size, void* d_ws, size_t ws_size,
                              hipStream_t stream) {
  const float* x     = (const float*)d_in[0];
  const float* wq    = (const float*)d_in[1];
  const float* bq    = (const float*)d_in[2];
  const float* wk    = (const float*)d_in[3];
  const float* bk    = (const float*)d_in[4];
  const float* wv    = (const float*)d_in[5];
  const float* bv    = (const float*)d_in[6];
  const float* gamma = (const float*)d_in[7];
  float* out = (float*)d_out;

  char* ws = (char*)d_ws;
  const size_t QT_BYTES = (size_t)NB * NVOX * QK_DIM * sizeof(float);   // 512000
  float*    qt = (float*)ws;
  float*    kt = (float*)(ws + QT_BYTES);
  _Float16* vh = (_Float16*)(ws + 2 * QT_BYTES);                        // 2 MB

  dim3 grid1((NVOX + 255) / 256, NB);
  qkv_proj_kernel<<<grid1, 256, 0, stream>>>(x, wq, bq, wk, bk, wv, bv, qt, kt, vh);

  dim3 grid2(NB * TILES_PER_B);            // 250 blocks * 4 waves = 1000 waves
  attn_stream_kernel<<<grid2, MBLOCK * 2, 0, stream>>>(qt, kt, vh, x, gamma, out);
}
